// KPConv_10960756539891
// MI455X (gfx1250) — compile-verified, matrix operand order checked by reading, not run on hardware
//
#include <hip/hip_runtime.h>

// ---------------- problem constants (match reference) ----------------
#define N_Q   50000
#define M_S   50000
#define H_NB  32
#define K_KP  15
#define CIN   64
#define COUT  128
#define KP_EXT 0.06f

#define QT 16                 // queries per WMMA M-tile
#define TILES 8               // query tiles per block (128 queries/block)
#define FPK_STRIDE_DW 492     // 960 bf16 = 480 dw + 12 pad dw (bank-spread)
#define NCHUNK 30             // 960 / 32 contraction chunks
#define NWAVE 8

typedef __attribute__((ext_vector_type(16))) __bf16 v16bf;
typedef __attribute__((ext_vector_type(8)))  float  v8f;
typedef __attribute__((ext_vector_type(4)))  float  v4f;
typedef __attribute__((ext_vector_type(2)))  float  v2f;

union BfVec { unsigned int w[8]; v16bf v; };

__device__ __forceinline__ unsigned int pack_bf16(float a, float b) {
  unsigned int ua = __float_as_uint(a);
  unsigned int ub = __float_as_uint(b);
  ua = (ua + 0x7FFFu + ((ua >> 16) & 1u)) >> 16;   // RNE
  ub = (ub + 0x7FFFu + ((ub >> 16) & 1u)) >> 16;
  return (ub << 16) | (ua & 0xFFFFu);
}

// ---- prepack: weights [960,128] f32 -> bf16 B-fragments in d_ws ----
// layout: fragment(chunk, wave, lane) = 8 dwords contiguous
__global__ __launch_bounds__(256) void prepack_w(const float* __restrict__ weights,
                                                 uint4* __restrict__ wsB) {
  const int tid = blockIdx.x * 256 + threadIdx.x;      // 0 .. 30*8*32-1
  if (tid >= NCHUNK * NWAVE * 32) return;
  const int l    = tid & 31;
  const int w    = (tid >> 5) & 7;
  const int c    = tid >> 8;
  const int o    = w * 16 + (l & 15);
  const int koff = (l >> 4) * 8;
  const int kc0  = c * 32;
  const float* w0 = weights + (size_t)(kc0 + koff) * COUT + o;
  const float* w1 = w0 + 16 * COUT;
  unsigned int d[8];
  #pragma unroll
  for (int e = 0; e < 8; e += 2) {
    d[e >> 1]       = pack_bf16(w0[(size_t)e * COUT], w0[(size_t)(e + 1) * COUT]);
    d[4 + (e >> 1)] = pack_bf16(w1[(size_t)e * COUT], w1[(size_t)(e + 1) * COUT]);
  }
  uint4* dst = wsB + (size_t)tid * 2;
  dst[0] = make_uint4(d[0], d[1], d[2], d[3]);
  dst[1] = make_uint4(d[4], d[5], d[6], d[7]);
}

template <int USE_WS>
__global__ __launch_bounds__(256) void kpconv_fused(
    const float* __restrict__ q_pts,          // [N,3]
    const float* __restrict__ s_pts,          // [M,3]
    const float* __restrict__ x,              // [M,CIN]
    const float* __restrict__ weights,        // [K,CIN,COUT] = [960,128]
    const float* __restrict__ kernel_points,  // [K,3]
    const int*   __restrict__ inds,           // [N,H]
    const uint4* __restrict__ wsB,            // prepacked B fragments (or null)
    float*       __restrict__ out)            // [N,COUT]
{
  __shared__ __align__(16) float        h_lds[QT][H_NB][16];               // 32 KB
  __shared__ __align__(16) unsigned int fpk_lds[TILES][QT][FPK_STRIDE_DW]; // ~246 KB
  __shared__               int          idx_lds[QT][H_NB];                 // 2 KB
  __shared__               float        kp_lds[K_KP * 3];

  const int t       = threadIdx.x;
  const int wave    = t >> 5;
  const int lane    = t & 31;
  const int blk_q0  = blockIdx.x * (QT * TILES);

  if (t < K_KP * 3) kp_lds[t] = kernel_points[t];

  // ---------------- per-tile: h, gather, fpk (bf16) ----------------
  for (int tile = 0; tile < TILES; ++tile) {
    const int q0 = blk_q0 + tile * QT;
    const bool tile_valid = (q0 < N_Q);        // N % 16 == 0 -> tiles full or absent
    __syncthreads();                           // kp ready / h_lds free for reuse

    if (tile_valid) {
      // Phase 1: kernel-point influences h[q][h][k] (f32), 2 (q,h) pairs/thread
      #pragma unroll
      for (int pp = 0; pp < 2; ++pp) {
        const int p  = t + pp * 256;
        const int q  = p >> 5;
        const int hh = p & 31;
        const int gq = q0 + q;
        const int idx = inds[gq * H_NB + hh];
        idx_lds[q][hh] = idx;
        float hv[16];
        if (idx < M_S) {
          const float rx = s_pts[idx * 3 + 0] - q_pts[gq * 3 + 0];
          const float ry = s_pts[idx * 3 + 1] - q_pts[gq * 3 + 1];
          const float rz = s_pts[idx * 3 + 2] - q_pts[gq * 3 + 2];
          #pragma unroll
          for (int k = 0; k < K_KP; ++k) {
            const float dx = rx - kp_lds[k * 3 + 0];
            const float dy = ry - kp_lds[k * 3 + 1];
            const float dz = rz - kp_lds[k * 3 + 2];
            const float d2 = fmaf(dx, dx, fmaf(dy, dy, dz * dz));
            hv[k] = fmaxf(1.0f - sqrtf(d2) * (1.0f / KP_EXT), 0.0f);
          }
        } else {
          #pragma unroll
          for (int k = 0; k < K_KP; ++k) hv[k] = 0.0f;   // shadow neighbor
        }
        hv[15] = 0.0f;
        float4* dst = (float4*)&h_lds[q][hh][0];
        dst[0] = make_float4(hv[0],  hv[1],  hv[2],  hv[3]);
        dst[1] = make_float4(hv[4],  hv[5],  hv[6],  hv[7]);
        dst[2] = make_float4(hv[8],  hv[9],  hv[10], hv[11]);
        dst[3] = make_float4(hv[12], hv[13], hv[14], hv[15]);
      }
    }
    __syncthreads();

    if (tile_valid) {
      // Phase 2: fpk[q][k][c4..c4+3] as v2f pairs -> V_PK_FMA_F32 path
      const int q  = t >> 4;
      const int c4 = (t & 15) * 4;
      v2f acc[K_KP][2];
      #pragma unroll
      for (int k = 0; k < K_KP; ++k) {
        acc[k][0] = (v2f)0.0f;
        acc[k][1] = (v2f)0.0f;
      }

      #pragma unroll 4
      for (int hh = 0; hh < H_NB; ++hh) {
        const int idx = idx_lds[q][hh];
        if (idx < M_S) {
          const v4f xv = *(const v4f*)(x + (size_t)idx * CIN + c4);
          const v2f xlo = {xv.x, xv.y};
          const v2f xhi = {xv.z, xv.w};
          const v4f* hr = (const v4f*)&h_lds[q][hh][0];
          const v4f ha = hr[0];
          const v4f hb = hr[1];
          const v4f hc = hr[2];
          const v4f hd = hr[3];
          const float hk[15] = {ha.x, ha.y, ha.z, ha.w, hb.x, hb.y, hb.z, hb.w,
                                hc.x, hc.y, hc.z, hc.w, hd.x, hd.y, hd.z};
          #pragma unroll
          for (int k = 0; k < K_KP; ++k) {
            const v2f hs = (v2f)hk[k];   // splat
            acc[k][0] = __builtin_elementwise_fma(hs, xlo, acc[k][0]);
            acc[k][1] = __builtin_elementwise_fma(hs, xhi, acc[k][1]);
          }
        }
      }
      // Phase 3: pack fpk row to bf16 (kc = k*64 + c)
      unsigned int* frow = &fpk_lds[tile][q][0];
      const int base = c4 >> 1;
      #pragma unroll
      for (int k = 0; k < K_KP; ++k) {
        frow[k * 32 + base]     = pack_bf16(acc[k][0].x, acc[k][0].y);
        frow[k * 32 + base + 1] = pack_bf16(acc[k][1].x, acc[k][1].y);
      }
    }
  }
  __syncthreads();   // all fpk tiles visible

  // ---------------- stage 2: [128,960] x [960,128] via bf16 WMMA ----------------
  // wave w -> output cols [16w, 16w+16); lanes 0-15: K 0-7/16-23, lanes 16-31: K 8-15/24-31
  const int col  = lane & 15;
  const int koff = (lane >> 4) * 8;
  const int o    = wave * 16 + col;
  const int qrow = lane & 15;
  const int nt   = min(TILES, (N_Q - blk_q0 + QT - 1) / QT);

  v8f accv[TILES];
  #pragma unroll
  for (int tt = 0; tt < TILES; ++tt) accv[tt] = (v8f){0, 0, 0, 0, 0, 0, 0, 0};

  #pragma unroll 2
  for (int chunk = 0; chunk < NCHUNK; ++chunk) {
    const int kc0 = chunk * 32;

    // B fragment (shared by all 8 tiles)
    BfVec b;
    if (USE_WS) {
      const uint4* bp = wsB + ((size_t)(chunk * NWAVE + wave) * 32 + lane) * 2;
      const uint4 b0 = bp[0];
      const uint4 b1 = bp[1];
      b.w[0] = b0.x; b.w[1] = b0.y; b.w[2] = b0.z; b.w[3] = b0.w;
      b.w[4] = b1.x; b.w[5] = b1.y; b.w[6] = b1.z; b.w[7] = b1.w;
    } else {
      const float* w0 = weights + (size_t)(kc0 + koff) * COUT + o;
      const float* w1 = w0 + 16 * COUT;
      #pragma unroll
      for (int e = 0; e < 8; e += 2) {
        b.w[e >> 1]       = pack_bf16(w0[(size_t)e * COUT], w0[(size_t)(e + 1) * COUT]);
        b.w[4 + (e >> 1)] = pack_bf16(w1[(size_t)e * COUT], w1[(size_t)(e + 1) * COUT]);
      }
    }

    // Prefetch all A fragments (branch-free), then burst the WMMAs
    BfVec a[TILES];
    #pragma unroll
    for (int tt = 0; tt < TILES; ++tt) {
      const unsigned int* arow = &fpk_lds[tt][qrow][0];
      const uint4 alo = *(const uint4*)(arow + (kc0 >> 1) + (koff >> 1));
      const uint4 ahi = *(const uint4*)(arow + ((kc0 + 16) >> 1) + (koff >> 1));
      a[tt].w[0] = alo.x; a[tt].w[1] = alo.y; a[tt].w[2] = alo.z; a[tt].w[3] = alo.w;
      a[tt].w[4] = ahi.x; a[tt].w[5] = ahi.y; a[tt].w[6] = ahi.z; a[tt].w[7] = ahi.w;
    }
    #pragma unroll
    for (int tt = 0; tt < TILES; ++tt) {
      accv[tt] = __builtin_amdgcn_wmma_f32_16x16x32_bf16(
          false, a[tt].v, false, b.v, (short)0, accv[tt], false, false);
    }
  }

  // store: VGPR r, lanes 0-15 -> M=r, lanes 16-31 -> M=r+8
  #pragma unroll
  for (int tt = 0; tt < TILES; ++tt) {
    if (tt < nt) {
      const int qbase = blk_q0 + tt * QT + (lane >> 4) * 8;
      #pragma unroll
      for (int r = 0; r < 8; ++r) {
        out[(size_t)(qbase + r) * COUT + o] = accv[tt][r];
      }
    }
  }
}

extern "C" void kernel_launch(void* const* d_in, const int* in_sizes, int n_in,
                              void* d_out, int out_size, void* d_ws, size_t ws_size,
                              hipStream_t stream) {
  const float* q_pts         = (const float*)d_in[0];
  const float* s_pts         = (const float*)d_in[1];
  const float* x             = (const float*)d_in[2];
  const float* weights       = (const float*)d_in[3];
  const float* kernel_points = (const float*)d_in[4];
  const int*   neighb_inds   = (const int*)d_in[5];
  float* out = (float*)d_out;

  const int    qpb       = QT * TILES;                       // 128 queries per block
  const int    grid      = (N_Q + qpb - 1) / qpb;            // 391
  const size_t wsb_bytes = (size_t)NCHUNK * NWAVE * 32 * 8 * 4;  // 245760

  if (ws_size >= wsb_bytes) {
    uint4* wsB = (uint4*)d_ws;
    hipLaunchKernelGGL(prepack_w, dim3(NCHUNK), dim3(256), 0, stream, weights, wsB);
    hipLaunchKernelGGL((kpconv_fused<1>), dim3(grid), dim3(256), 0, stream,
                       q_pts, s_pts, x, weights, kernel_points, neighb_inds,
                       (const uint4*)wsB, out);
  } else {
    hipLaunchKernelGGL((kpconv_fused<0>), dim3(grid), dim3(256), 0, stream,
                       q_pts, s_pts, x, weights, kernel_points, neighb_inds,
                       (const uint4*)nullptr, out);
  }
}
